// RobustFill_57475252355555
// MI455X (gfx1250) — matrix operand order repaired and checked by reading, not compiled, not run on hardware
//
#include <hip/hip_runtime.h>

// ---------------- problem constants ----------------
#define Vv   128
#define V1C  129
#define Hh   512
#define G4   2048      // 4*H
#define Ee   128
#define NEXC 4
#define Ll   32
#define Bb   512
#define Tt   32
#define RR   2048      // NEX*B rows

// ---------------- types ----------------
typedef __attribute__((ext_vector_type(16))) __bf16 v16bf;
typedef __attribute__((ext_vector_type(8)))  float  v8f;
typedef __attribute__((ext_vector_type(4)))  unsigned int v4u;

union FragBF { v16bf v; v4u q[2]; };

__device__ __forceinline__ unsigned short f2bf(float f) {
    unsigned int u = __float_as_uint(f);
    u += 0x7fffu + ((u >> 16) & 1u);          // round-to-nearest-even
    return (unsigned short)(u >> 16);
}
__device__ __forceinline__ float bf2f(unsigned short h) {
    return __uint_as_float(((unsigned int)h) << 16);
}
__device__ __forceinline__ float sigf(float x) { return 1.0f / (1.0f + __expf(-x)); }

// ================================================================
// bf16 GEMM (small tile):  C[M][N](f32) = A[M][K] * B[N][K]^T
// 128 threads = 4 waves; 64x64 block tile; 2x2 WMMA per wave.
// Used for narrow-N GEMMs (u, fc, logits) where parallelism matters.
// ================================================================
__global__ __launch_bounds__(128)
void gemm_bf16(const unsigned short* __restrict__ A, int lda,
               const unsigned short* __restrict__ B, int ldb,
               float* __restrict__ C, int ldc, int N, int K)
{
    const int lane = threadIdx.x & 31;
    const int wave = threadIdx.x >> 5;
    const int wm = wave >> 1, wn = wave & 1;
    const int mBase = blockIdx.y * 64 + wm * 32;
    const int nBase = blockIdx.x * 64 + wn * 32;
    const int l15 = lane & 15;
    const int hi  = lane >> 4;

    v8f c00 = {}, c01 = {}, c10 = {}, c11 = {};

    // A fragment (16x32 bf16): lane row = l15, K runs [k+8*hi, +8) and [k+16+8*hi, +8)
    const unsigned short* a0p = A + (size_t)(mBase + l15     ) * lda + hi * 8;
    const unsigned short* a1p = A + (size_t)(mBase + 16 + l15) * lda + hi * 8;
    // B fragment (32x16 bf16): lane col = l15, K runs [k+8*hi, +8) and [k+16+8*hi, +8)
    const unsigned short* b0p = B + (size_t)(nBase + l15     ) * ldb + hi * 8;
    const unsigned short* b1p = B + (size_t)(nBase + 16 + l15) * ldb + hi * 8;

    for (int k = 0; k < K; k += 32) {
        FragBF a0, a1, b0, b1;
        a0.q[0] = *(const v4u*)(a0p + k);     a0.q[1] = *(const v4u*)(a0p + k + 16);
        a1.q[0] = *(const v4u*)(a1p + k);     a1.q[1] = *(const v4u*)(a1p + k + 16);
        b0.q[0] = *(const v4u*)(b0p + k);     b0.q[1] = *(const v4u*)(b0p + k + 16);
        b1.q[0] = *(const v4u*)(b1p + k);     b1.q[1] = *(const v4u*)(b1p + k + 16);
        c00 = __builtin_amdgcn_wmma_f32_16x16x32_bf16(false, a0.v, false, b0.v, (short)0, c00, false, false);
        c01 = __builtin_amdgcn_wmma_f32_16x16x32_bf16(false, a0.v, false, b1.v, (short)0, c01, false, false);
        c10 = __builtin_amdgcn_wmma_f32_16x16x32_bf16(false, a1.v, false, b0.v, (short)0, c10, false, false);
        c11 = __builtin_amdgcn_wmma_f32_16x16x32_bf16(false, a1.v, false, b1.v, (short)0, c11, false, false);
    }
    // C/D layout: v8f element r at (row = base + r + 8*hi, col = base + l15)
    #pragma unroll
    for (int r = 0; r < 8; ++r) {
        int row0 = mBase + r + 8 * hi;
        int col0 = nBase + l15;
        if (col0      < N) C[(size_t)row0        * ldc + col0     ] = c00[r];
        if (col0 + 16 < N) C[(size_t)row0        * ldc + col0 + 16] = c01[r];
        if (col0      < N) C[(size_t)(row0 + 16) * ldc + col0     ] = c10[r];
        if (col0 + 16 < N) C[(size_t)(row0 + 16) * ldc + col0 + 16] = c11[r];
    }
}

// ================================================================
// bf16 GEMM (big tile): 128 threads = 4 waves arranged 2x2; 128x128
// block tile; 4x4 WMMA per wave (64x64). 16 b128 loads per 16 WMMAs
// -> 0.5 KB L2 traffic per WMMA (half of the 2x2 variant). Used for
// the dominant N=2048 gate GEMMs. Requires M%128==0, N%128==0, K%32==0.
// ================================================================
__global__ __launch_bounds__(128)
void gemm_bf16_big(const unsigned short* __restrict__ A, int lda,
                   const unsigned short* __restrict__ B, int ldb,
                   float* __restrict__ C, int ldc, int K)
{
    const int lane = threadIdx.x & 31;
    const int wave = threadIdx.x >> 5;
    const int wm = wave >> 1, wn = wave & 1;
    const int mBase = blockIdx.y * 128 + wm * 64;
    const int nBase = blockIdx.x * 128 + wn * 64;
    const int l15 = lane & 15;
    const int hi  = lane >> 4;

    v8f acc[16];
    #pragma unroll
    for (int i = 0; i < 16; ++i) acc[i] = (v8f){};

    const unsigned short* ap[4];
    const unsigned short* bp[4];
    #pragma unroll
    for (int i = 0; i < 4; ++i) {
        ap[i] = A + (size_t)(mBase + 16 * i + l15) * lda + hi * 8;
        bp[i] = B + (size_t)(nBase + 16 * i + l15) * ldb + hi * 8;
    }

    for (int k = 0; k < K; k += 32) {
        FragBF a[4], b[4];
        #pragma unroll
        for (int i = 0; i < 4; ++i) {
            a[i].q[0] = *(const v4u*)(ap[i] + k);
            a[i].q[1] = *(const v4u*)(ap[i] + k + 16);
        }
        #pragma unroll
        for (int i = 0; i < 4; ++i) {
            b[i].q[0] = *(const v4u*)(bp[i] + k);
            b[i].q[1] = *(const v4u*)(bp[i] + k + 16);
        }
        #pragma unroll
        for (int i = 0; i < 4; ++i)
            #pragma unroll
            for (int jj = 0; jj < 4; ++jj)
                acc[i * 4 + jj] = __builtin_amdgcn_wmma_f32_16x16x32_bf16(
                    false, a[i].v, false, b[jj].v, (short)0, acc[i * 4 + jj], false, false);
    }

    #pragma unroll
    for (int i = 0; i < 4; ++i) {
        #pragma unroll
        for (int jj = 0; jj < 4; ++jj) {
            int col = nBase + 16 * jj + l15;
            #pragma unroll
            for (int r = 0; r < 8; ++r) {
                int row = mBase + 16 * i + r + 8 * hi;
                C[(size_t)row * ldc + col] = acc[i * 4 + jj][r];
            }
        }
    }
}

// ---------------- weight prep ----------------
__global__ void conv_bf16_kernel(const float* __restrict__ src, unsigned short* __restrict__ dst, int n) {
    int i = blockIdx.x * 256 + threadIdx.x;
    if (i < n) dst[i] = f2bf(src[i]);
}
// V_w [129][128] -> zero-padded bf16 [192][128] so the 3rd 64-wide N tile has valid B rows
__global__ void conv_vw_kernel(const float* __restrict__ Vw, unsigned short* __restrict__ dst) {
    int i = blockIdx.x * 256 + threadIdx.x;
    if (i >= 192 * Ee) return;
    int v = i >> 7;
    dst[i] = (v < V1C) ? f2bf(Vw[i]) : (unsigned short)0;
}
// gate table: tab[v][n] = Wih[n][v] + bih[n] + bhh[n]   (one-hot input contribution)
__global__ void build_tab_kernel(const float* __restrict__ Wih, const float* __restrict__ bih,
                                 const float* __restrict__ bhh, float* __restrict__ tab) {
    int i = blockIdx.x * 256 + threadIdx.x;
    if (i >= V1C * G4) return;
    int v = i / G4, n = i % G4;
    tab[i] = Wih[(size_t)n * V1C + v] + bih[n] + bhh[n];
}

// ---------------- state init ----------------
__global__ void init_state_kernel(const float* __restrict__ h0, const float* __restrict__ c0,
                                  unsigned short* __restrict__ hbuf, float* __restrict__ cbuf) {
    int i = blockIdx.x * 256 + threadIdx.x;
    if (i >= RR * Hh) return;
    int n = i % Hh;
    hbuf[i] = f2bf(h0[n]);
    cbuf[i] = c0[n];
}
__global__ void init_score_kernel(float* __restrict__ score, float* __restrict__ act) {
    int b = blockIdx.x * 256 + threadIdx.x;
    if (b < Bb) { score[b] = 0.0f; act[b] = 1.0f; }
}

// ---------------- encoder LSTM pointwise ----------------
__global__ void lstm_enc_point(const float* __restrict__ gates, const float* __restrict__ tab,
                               const int* __restrict__ inputs, int t,
                               float* __restrict__ cbuf, unsigned short* __restrict__ hbuf,
                               unsigned short* __restrict__ Hs) {
    int i = blockIdx.x * 256 + threadIdx.x;
    if (i >= RR * Hh) return;
    int r = i / Hh, n = i % Hh;
    int j = r / Bb, b = r % Bb;
    int tok = inputs[((size_t)j * Ll + t) * Bb + b];
    const float* tr = tab + (size_t)tok * G4;
    const float* gr = gates + (size_t)r * G4;
    float gi = gr[n]            + tr[n];
    float gf = gr[Hh + n]       + tr[Hh + n];
    float gg = gr[2 * Hh + n]   + tr[2 * Hh + n];
    float go = gr[3 * Hh + n]   + tr[3 * Hh + n];
    float c  = cbuf[i];
    float c2 = sigf(gf) * c + sigf(gi) * tanhf(gg);
    float h2 = sigf(go) * tanhf(c2);
    cbuf[i] = c2;
    unsigned short hb = f2bf(h2);
    hbuf[i] = hb;
    Hs[((size_t)(j * Ll + t) * Bb + b) * Hh + n] = hb;
}

// ---------------- decoder LSTM pointwise (h written into hcat left half, ld=1024) ----------------
__global__ void lstm_dec_point(const float* __restrict__ gates, const float* __restrict__ tab,
                               const int* __restrict__ target, int t,
                               const float* __restrict__ c_in, int c_bcast,
                               float* __restrict__ c_out, unsigned short* __restrict__ hcat) {
    int i = blockIdx.x * 256 + threadIdx.x;
    if (i >= RR * Hh) return;
    int r = i / Hh, n = i % Hh;
    int b = r % Bb;
    int tok = (t < 0) ? Vv : target[(size_t)t * Bb + b];   // SOS one-hot index == V
    const float* tr = tab + (size_t)tok * G4;
    const float* gr = gates + (size_t)r * G4;
    float gi = gr[n]          + tr[n];
    float gf = gr[Hh + n]     + tr[Hh + n];
    float gg = gr[2 * Hh + n] + tr[2 * Hh + n];
    float go = gr[3 * Hh + n] + tr[3 * Hh + n];
    float c  = c_bcast ? c_in[n] : c_in[i];
    float c2 = sigf(gf) * c + sigf(gi) * tanhf(gg);
    c_out[i] = c2;
    hcat[(size_t)r * 1024 + n] = f2bf(sigf(go) * tanhf(c2));
}

// ---------------- embedding gather + first-invalid index ----------------
__global__ void emb_gather_kernel(const int* __restrict__ inputs, const unsigned short* __restrict__ Hs,
                                  unsigned short* __restrict__ hcat, int* __restrict__ fbuf) {
    int p = blockIdx.x * 256 + threadIdx.x;
    if (p >= RR) return;
    int j = p / Bb, b = p % Bb;
    int f = Ll;
    for (int t = 0; t < Ll; ++t) {
        if (inputs[((size_t)j * Ll + t) * Bb + b] == Vv) { f = t; break; }
    }
    fbuf[p] = f;                         // attention position l valid iff l <= f
    int idx = (f < Ll) ? f : (Ll - 1);   // emb index = min(f, L-1)
    const v4u* src = (const v4u*)(Hs + ((size_t)(j * Ll + idx) * Bb + b) * Hh);
    v4u* dst = (v4u*)(hcat + (size_t)p * 1024);
    for (int q = 0; q < Hh / 8; ++q) dst[q] = src[q];
}

// ---------------- attention: scores -> masked softmax over L -> ctx (one wave per (j,b)) ----------------
__global__ __launch_bounds__(128)
void attention_kernel(const unsigned short* __restrict__ Hs, const float* __restrict__ u,
                      const int* __restrict__ fbuf, unsigned short* __restrict__ hcat) {
    int lane = threadIdx.x & 31;
    int wave = threadIdx.x >> 5;
    int p = blockIdx.x * 4 + wave;       // (j,b) pair
    int j = p / Bb, b = p % Bb;

    // score for l = lane: dot(Hs[j][l][b][:], u[p][:])
    const unsigned short* hp = Hs + ((size_t)(j * Ll + lane) * Bb + b) * Hh;
    const float* up = u + (size_t)p * Hh;
    float s = 0.0f;
    for (int h = 0; h < Hh; h += 8) {
        v4u q = *(const v4u*)(hp + h);
        #pragma unroll
        for (int e = 0; e < 4; ++e) {
            unsigned int w = q[e];
            s += bf2f((unsigned short)(w & 0xffffu)) * up[h + 2 * e];
            s += bf2f((unsigned short)(w >> 16))     * up[h + 2 * e + 1];
        }
    }
    int f = fbuf[p];
    if (lane > f) s = -1e9f;
    float m = s;
    #pragma unroll
    for (int o = 16; o > 0; o >>= 1) m = fmaxf(m, __shfl_xor(m, o));
    float ex = __expf(s - m);
    float sum = ex;
    #pragma unroll
    for (int o = 16; o > 0; o >>= 1) sum += __shfl_xor(sum, o);
    float a = ex / sum;

    // ctx[hb..hb+16) with hb = lane*16; broadcast a_l via shuffles
    float acc[16];
    #pragma unroll
    for (int e = 0; e < 16; ++e) acc[e] = 0.0f;
    int hb = lane * 16;
    for (int ll = 0; ll < Ll; ++ll) {
        float all = __shfl(a, ll);
        const unsigned short* hp2 = Hs + ((size_t)(j * Ll + ll) * Bb + b) * Hh + hb;
        v4u q0 = *(const v4u*)hp2;
        v4u q1 = *(const v4u*)(hp2 + 8);
        #pragma unroll
        for (int e = 0; e < 4; ++e) {
            unsigned int w0 = q0[e], w1 = q1[e];
            acc[2 * e]     += all * bf2f((unsigned short)(w0 & 0xffffu));
            acc[2 * e + 1] += all * bf2f((unsigned short)(w0 >> 16));
            acc[8 + 2 * e] += all * bf2f((unsigned short)(w1 & 0xffffu));
            acc[9 + 2 * e] += all * bf2f((unsigned short)(w1 >> 16));
        }
    }
    v4u o0, o1;
    #pragma unroll
    for (int e = 0; e < 4; ++e) {
        o0[e] = (unsigned int)f2bf(acc[2 * e])     | ((unsigned int)f2bf(acc[2 * e + 1]) << 16);
        o1[e] = (unsigned int)f2bf(acc[8 + 2 * e]) | ((unsigned int)f2bf(acc[9 + 2 * e]) << 16);
    }
    unsigned short* op = hcat + (size_t)p * 1024 + Hh + hb;   // ctx into right half of concat
    *(v4u*)op = o0;
    *(v4u*)(op + 8) = o1;
}

// ---------------- tanh(fc + W_b), max over examples, -> bf16 ----------------
__global__ void maxtanh_kernel(const float* __restrict__ fcraw, const float* __restrict__ Wb,
                               unsigned short* __restrict__ mbf) {
    int i = blockIdx.x * 256 + threadIdx.x;
    if (i >= Bb * Ee) return;
    int b = i / Ee, e = i % Ee;
    float m = -1e30f;
    #pragma unroll
    for (int j = 0; j < NEXC; ++j)
        m = fmaxf(m, tanhf(fcraw[(size_t)(j * Bb + b) * Ee + e] + Wb[e]));
    mbf[i] = f2bf(m);
}

// ---------------- log_softmax over V1, gather target, score/act update (one wave per b) ----------------
__global__ __launch_bounds__(128)
void score_kernel(const float* __restrict__ logits, const float* __restrict__ Vb,
                  const int* __restrict__ target, int t,
                  float* __restrict__ score, float* __restrict__ act) {
    int lane = threadIdx.x & 31;
    int wave = threadIdx.x >> 5;
    int b = blockIdx.x * 4 + wave;
    float v[5];
    #pragma unroll
    for (int i = 0; i < 5; ++i) {
        int idx = lane + 32 * i;
        v[i] = (idx < V1C) ? (logits[(size_t)b * V1C + idx] + Vb[idx]) : -1e30f;
    }
    float m = v[0];
    #pragma unroll
    for (int i = 1; i < 5; ++i) m = fmaxf(m, v[i]);
    #pragma unroll
    for (int o = 16; o > 0; o >>= 1) m = fmaxf(m, __shfl_xor(m, o));
    float s = 0.0f;
    #pragma unroll
    for (int i = 0; i < 5; ++i) s += (lane + 32 * i < V1C) ? __expf(v[i] - m) : 0.0f;
    #pragma unroll
    for (int o = 16; o > 0; o >>= 1) s += __shfl_xor(s, o);
    float logZ = m + __logf(s);
    if (lane == 0) {
        int tok = target[(size_t)t * Bb + b];
        float chosen = logits[(size_t)b * V1C + tok] + Vb[tok] - logZ;
        float a = act[b];
        score[b] += chosen * a;
        act[b] = a * ((tok != Vv) ? 1.0f : 0.0f);
    }
}

// ================================================================
extern "C" void kernel_launch(void* const* d_in, const int* in_sizes, int n_in,
                              void* d_out, int out_size, void* d_ws, size_t ws_size,
                              hipStream_t stream) {
    (void)in_sizes; (void)n_in; (void)out_size; (void)ws_size;

    const int*   inputs  = (const int*)  d_in[0];   // [NEX,L,B]
    const int*   target  = (const int*)  d_in[1];   // [T,B]
    const float* enc_Wih = (const float*)d_in[2];
    const float* enc_Whh = (const float*)d_in[3];
    const float* enc_bih = (const float*)d_in[4];
    const float* enc_bhh = (const float*)d_in[5];
    const float* enc_h0  = (const float*)d_in[6];
    const float* enc_c0  = (const float*)d_in[7];
    const float* dec_Wih = (const float*)d_in[8];
    const float* dec_Whh = (const float*)d_in[9];
    const float* dec_bih = (const float*)d_in[10];
    const float* dec_bhh = (const float*)d_in[11];
    const float* dec_c0  = (const float*)d_in[12];
    const float* W_w     = (const float*)d_in[13];
    const float* W_b     = (const float*)d_in[14];
    const float* V_w     = (const float*)d_in[15];
    const float* V_b     = (const float*)d_in[16];
    const float* A_w     = (const float*)d_in[17];
    float*       score   = (float*)d_out;           // [B]

    // ---- workspace layout ----
    char* ws = (char*)d_ws;
    size_t off = 0;
    auto alloc = [&](size_t bytes) -> void* {
        void* p = ws + off;
        off = (off + bytes + 255) & ~(size_t)255;
        return p;
    };
    unsigned short* Hs       = (unsigned short*)alloc((size_t)NEXC * Ll * Bb * Hh * 2); // 64 MB, L2-resident
    float*          gates    = (float*)alloc((size_t)RR * G4 * 4);                      // 16 MB
    unsigned short* encWhh_b = (unsigned short*)alloc((size_t)G4 * Hh * 2);
    unsigned short* decWhh_b = (unsigned short*)alloc((size_t)G4 * Hh * 2);
    unsigned short* Aw_b     = (unsigned short*)alloc((size_t)Hh * Hh * 2);
    unsigned short* Ww_b     = (unsigned short*)alloc((size_t)Ee * 2 * Hh * 2);
    unsigned short* Vw_b     = (unsigned short*)alloc((size_t)192 * Ee * 2);            // padded to 3 N-tiles
    float*          enc_tab  = (float*)alloc((size_t)V1C * G4 * 4);
    float*          dec_tab  = (float*)alloc((size_t)V1C * G4 * 4);
    unsigned short* hbuf     = (unsigned short*)alloc((size_t)RR * Hh * 2);             // encoder h (ld 512)
    float*          cbuf     = (float*)alloc((size_t)RR * Hh * 4);                      // cell state (f32)
    unsigned short* hcat     = (unsigned short*)alloc((size_t)RR * 1024 * 2);           // [h | ctx], ld 1024
    float*          u        = (float*)alloc((size_t)RR * Hh * 4);
    float*          fcraw    = (float*)alloc((size_t)RR * Ee * 4);
    unsigned short* mbf      = (unsigned short*)alloc((size_t)Bb * Ee * 2);
    float*          logits   = (float*)alloc((size_t)Bb * V1C * 4);
    int*            fbuf     = (int*)alloc((size_t)RR * 4);
    float*          act      = (float*)alloc((size_t)Bb * 4);

    const dim3 blk128(128);
    auto g256 = [](int n) { return dim3((n + 255) / 256); };

    // ---- weight prep (f32 -> bf16; gate tables absorb one-hot matmuls + both biases) ----
    conv_bf16_kernel<<<g256(G4 * Hh), 256, 0, stream>>>(enc_Whh, encWhh_b, G4 * Hh);
    conv_bf16_kernel<<<g256(G4 * Hh), 256, 0, stream>>>(dec_Whh, decWhh_b, G4 * Hh);
    conv_bf16_kernel<<<g256(Hh * Hh), 256, 0, stream>>>(A_w, Aw_b, Hh * Hh);
    conv_bf16_kernel<<<g256(Ee * 2 * Hh), 256, 0, stream>>>(W_w, Ww_b, Ee * 2 * Hh);
    conv_vw_kernel<<<g256(192 * Ee), 256, 0, stream>>>(V_w, Vw_b);
    build_tab_kernel<<<g256(V1C * G4), 256, 0, stream>>>(enc_Wih, enc_bih, enc_bhh, enc_tab);
    build_tab_kernel<<<g256(V1C * G4), 256, 0, stream>>>(dec_Wih, dec_bih, dec_bhh, dec_tab);
    init_state_kernel<<<g256(RR * Hh), 256, 0, stream>>>(enc_h0, enc_c0, hbuf, cbuf);
    init_score_kernel<<<g256(Bb), 256, 0, stream>>>(score, act);

    // ---- encoder: 32 steps of (gates = h @ Whh^T) + pointwise, all 4 examples batched ----
    for (int t = 0; t < Ll; ++t) {
        gemm_bf16_big<<<dim3(G4 / 128, RR / 128), blk128, 0, stream>>>(hbuf, Hh, encWhh_b, Hh,
                                                                       gates, G4, Hh);
        lstm_enc_point<<<g256(RR * Hh), 256, 0, stream>>>(gates, enc_tab, inputs, t, cbuf, hbuf, Hs);
    }

    // ---- embedding gather + attention masks; decoder init step on SOS (c from dec_c0) ----
    emb_gather_kernel<<<g256(RR), 256, 0, stream>>>(inputs, Hs, hcat, fbuf);
    gemm_bf16_big<<<dim3(G4 / 128, RR / 128), blk128, 0, stream>>>(hcat, 1024, decWhh_b, Hh,
                                                                   gates, G4, Hh);
    lstm_dec_point<<<g256(RR * Hh), 256, 0, stream>>>(gates, dec_tab, target, -1,
                                                      dec_c0, 1, cbuf, hcat);

    // ---- decoder: 32 steps ----
    for (int t = 0; t < Tt; ++t) {
        // u = h @ A_w^T  (narrow N -> small-tile GEMM for parallelism)
        gemm_bf16<<<dim3(Hh / 64, RR / 64), blk128, 0, stream>>>(hcat, 1024, Aw_b, Hh,
                                                                 u, Hh, Hh, Hh);
        // masked softmax over L + ctx -> right half of hcat
        attention_kernel<<<dim3(RR / 4), blk128, 0, stream>>>(Hs, u, fbuf, hcat);
        // fc = [h|ctx] @ W_w^T
        gemm_bf16<<<dim3(Ee / 64, RR / 64), blk128, 0, stream>>>(hcat, 1024, Ww_b, 2 * Hh,
                                                                 fcraw, Ee, Ee, 2 * Hh);
        // m = max_j tanh(fc + W_b)
        maxtanh_kernel<<<g256(Bb * Ee), 256, 0, stream>>>(fcraw, W_b, mbf);
        // logits = m @ V_w^T  (N=129 guarded; V_w padded to 192 rows)
        gemm_bf16<<<dim3(3, Bb / 64), blk128, 0, stream>>>(mbf, Ee, Vw_b, Ee,
                                                           logits, V1C, V1C, Ee);
        // log_softmax + target gather + score/act update
        score_kernel<<<dim3(Bb / 4), blk128, 0, stream>>>(logits, V_b, target, t, score, act);
        // LSTM update: gates = h @ dec_Whh^T (dominant GEMM -> big tile)
        gemm_bf16_big<<<dim3(G4 / 128, RR / 128), blk128, 0, stream>>>(hcat, 1024, decWhh_b, Hh,
                                                                       gates, G4, Hh);
        lstm_dec_point<<<g256(RR * Hh), 256, 0, stream>>>(gates, dec_tab, target, t,
                                                          cbuf, 0, cbuf, hcat);
    }
}
